// SuperPointNet_52329881534733
// MI455X (gfx1250) — matrix operand order repaired and checked by reading, compiled, not run on hardware
//
#include <hip/hip_runtime.h>
#include <math.h>

typedef _Float16 f16;
typedef __attribute__((ext_vector_type(16))) _Float16 v16h;
typedef __attribute__((ext_vector_type(8)))  float    v8f;
typedef __attribute__((ext_vector_type(4)))  unsigned u32x4;
typedef __attribute__((ext_vector_type(8)))  int      i32x8;
typedef __attribute__((ext_vector_type(4)))  int      i32x4;

#define NSC (512*512)

struct TopkState {
  unsigned prefix;
  int kth;
  unsigned G, c1;
  unsigned hist[256];
};

// ---------------------------------------------------------------------------
// Weight packing: OIHW f32 -> [K/2][Npad] dwords of packed f16 pairs
// B[k][n] = w[n][cin][dy][dx], k = (dy*kk+dx)*Cin + cin ; n >= Cout -> 0
// ---------------------------------------------------------------------------
__global__ void pack_w(const float* __restrict__ w, unsigned* __restrict__ bp,
                       int Cin, int Cout, int kk, int Npad, int total) {
  int i = blockIdx.x * blockDim.x + threadIdx.x;
  if (i >= total) return;
  int n  = i % Npad;
  int kp = i / Npad;
  float v0 = 0.f, v1 = 0.f;
  if (n < Cout) {
    int k0 = kp * 2;
    int cell0 = k0 / Cin,       cin0 = k0 - cell0 * Cin;
    int cell1 = (k0 + 1) / Cin, cin1 = (k0 + 1) - cell1 * Cin;
    int dy0 = cell0 / kk, dx0 = cell0 - dy0 * kk;
    int dy1 = cell1 / kk, dx1 = cell1 - dy1 * kk;
    v0 = w[(((size_t)n * Cin + cin0) * kk + dy0) * kk + dx0];
    v1 = w[(((size_t)n * Cin + cin1) * kk + dy1) * kk + dx1];
  }
  union { f16 h[2]; unsigned u; } p;
  p.h[0] = (f16)v0; p.h[1] = (f16)v1;
  bp[i] = p.u;
}

// ---------------------------------------------------------------------------
// Layer 1a: direct conv, Cin=1 (K=9, too small for WMMA). img f32 -> f16 NHWC
// ---------------------------------------------------------------------------
__global__ void conv1a(const float* __restrict__ img, const float* __restrict__ w,
                       const float* __restrict__ bias, f16* __restrict__ out) {
  int i = blockIdx.x * blockDim.x + threadIdx.x;          // 2*512*512*64
  int n = i & 63;
  int t = i >> 6;
  int x = t & 511; t >>= 9;
  int y = t & 511;
  int b = t >> 9;
  float s = bias[n];
  #pragma unroll
  for (int dy = 0; dy < 3; ++dy)
    #pragma unroll
    for (int dx = 0; dx < 3; ++dx) {
      int yy = y + dy - 1, xx = x + dx - 1;
      if (yy >= 0 && yy < 512 && xx >= 0 && xx < 512)
        s += img[((size_t)b * 512 + yy) * 512 + xx] * w[(n * 3 + dy) * 3 + dx];
    }
  out[(size_t)i] = (f16)fmaxf(s, 0.0f);
}

// ---------------------------------------------------------------------------
// Implicit-GEMM conv, WMMA f16 16x16x32, f32 accumulate.
// Block = 8 waves sharing one 64-wide N tile; each wave owns a 32(M)x64(N)
// output tile (2 A frags x 4 B frags = 8 WMMAs per 32-K chunk).
// Per 3x3 cell, the B slice ((Cin/2) x 64 dwords, <= 32KB) is staged into LDS
// once per block via the Tensor Data Mover (TENSORcnt), then all 8 waves read
// B fragments from LDS. Requires W%32==0, Npad%64==0, Cin%32==0, tilesM%8==0.
// Grids are exact: no early returns, EXEC all-ones at every WMMA/barrier.
// ---------------------------------------------------------------------------
__global__ __launch_bounds__(256) void conv_wmma_big(
    const f16* __restrict__ act, const unsigned* __restrict__ bp,
    const float* __restrict__ bias,
    int Hh, int Ww, int Cin, int Npad, int kk, int relu,
    f16* __restrict__ out16, float* __restrict__ out32) {
  __shared__ unsigned ldsB[8192];                 // 32KB: up to (256/2) x 64
  const int lane   = threadIdx.x & 31;
  const int waveId = threadIdx.x >> 5;
  const int M = 2 * Hh * Ww;
  const int tilesM = M >> 5;
  const int blkPerN = tilesM >> 3;                // tilesM / 8 (exact)
  const int tileN = blockIdx.x / blkPerN;
  const int tileM = (blockIdx.x - tileN * blkPerN) * 8 + waveId;
  const int m0 = tileM << 5;
  const int n0 = tileN << 6;
  const int hw = Hh * Ww;
  const int bimg = m0 / hw;
  const int rem  = m0 - bimg * hw;
  const int y  = rem / Ww;
  const int x0 = rem - y * Ww;
  const int pad = kk >> 1;
  const int r  = lane & 15;
  const int hi = lane >> 4;
  const unsigned* act32 = (const unsigned*)act;

  // lane-invariant K-pair offsets within a 32-K chunk (contiguous: lanes 0-15 pairs 0..7, 16-31 pairs 8..15)
  int pairA[8];
  #pragma unroll
  for (int v = 0; v < 8; ++v)
    pairA[v] = v + (hi ? 8 : 0);
  const int pairB0 = hi ? 8 : 0;      // B: lanes 0-15 pairs 0..7, 16-31 pairs 8..15

  v8f acc[2][4] = {};
  const int nCb = Cin >> 5;           // 32-K chunks per cell
  const int nCells = kk * kk;
  const int nPairsCell = Cin >> 1;    // B rows (K pairs) per cell

  for (int cell = 0; cell < nCells; ++cell) {
    const int dy = cell / kk;
    const int dx = cell - dy * kk;
    const int yy = y + dy - pad;
    const bool yok = (yy >= 0) && (yy < Hh);
    const int xA0 = x0 + r + dx - pad;      // pixel x for A rows r and r+16
    const int xA1 = xA0 + 16;
    const size_t rowBase = ((size_t)bimg * Hh + (yok ? yy : 0)) * Ww;
    const bool ok0 = yok && (xA0 >= 0) && (xA0 < Ww);
    const bool ok1 = yok && (xA1 >= 0) && (xA1 < Ww);
    const size_t aBase0 = (rowBase + (size_t)(ok0 ? xA0 : 0)) * Cin;
    const size_t aBase1 = (rowBase + (size_t)(ok1 ? xA1 : 0)) * Cin;
    const int cellPairBase = (cell * Cin) >> 1;

    __syncthreads();                         // previous cell's LDS reads done
#if __has_builtin(__builtin_amdgcn_tensor_load_to_lds)
    if (threadIdx.x < 32) {
      // TDM D#: 2D tile, 64 (dim0) x nPairsCell (dim1) dwords,
      // row stride Npad dwords, from bp[cellPairBase][n0] into ldsB.
      unsigned long long gaddr =
          (unsigned long long)(const void*)(bp + (size_t)cellPairBase * Npad + n0);
      unsigned ldsOff = (unsigned)(size_t)(void*)&ldsB[0];
      u32x4 g0 = (u32x4)0;
      g0.x = 1u;                                             // count=1, load
      g0.y = ldsOff;                                         // lds_addr
      g0.z = (unsigned)gaddr;                                // global_addr[31:0]
      g0.w = (unsigned)((gaddr >> 32) & 0x1FFFFFFull) | (2u << 30); // [56:32]|type=2
      i32x8 g1 = (i32x8)0;
      g1[0] = (int)(2u << 16);                               // data_size = 4B
      g1[1] = (int)(((unsigned)Npad & 0xFFFFu) << 16);       // tensor_dim0 lo16
      g1[2] = (int)((((unsigned)Npad >> 16) & 0xFFFFu) |
                    (((unsigned)nPairsCell & 0xFFFFu) << 16)); // td0 hi | td1 lo
      g1[3] = (int)(64u << 16);                              // td1 hi=0 | tile_dim0=64
      g1[4] = (int)((unsigned)nPairsCell & 0xFFFFu);         // tile_dim1 | tile_dim2=0
      g1[5] = (int)(unsigned)Npad;                           // tensor_dim0_stride lo32
      g1[6] = 0;
      g1[7] = 0;
      i32x4 gz = (i32x4)0;
#if defined(__clang_major__) && (__clang_major__ >= 23)
      i32x8 gz8 = (i32x8)0;
      __builtin_amdgcn_tensor_load_to_lds(g0, g1, gz, gz, gz8, 0);
#else
      __builtin_amdgcn_tensor_load_to_lds(g0, g1, gz, gz, 0);
#endif
      __builtin_amdgcn_s_wait_tensorcnt(0);
    }
#else
    for (int e = threadIdx.x; e < nPairsCell * 64; e += 256) {
      int pr = e >> 6, col = e & 63;
      ldsB[e] = bp[(size_t)(cellPairBase + pr) * Npad + n0 + col];
    }
#endif
    __syncthreads();                         // B slice visible to all waves

    for (int cb = 0; cb < nCb; ++cb) {
      const int cinBase = cb << 5;
      union { unsigned u[8]; v16h h; } A0, A1;
      #pragma unroll
      for (int v = 0; v < 8; ++v) {
        unsigned a0 = 0, a1 = 0;
        if (ok0) a0 = act32[((aBase0 + cinBase) >> 1) + pairA[v]];
        if (ok1) a1 = act32[((aBase1 + cinBase) >> 1) + pairA[v]];
        A0.u[v] = a0; A1.u[v] = a1;
      }
      #pragma unroll
      for (int nb = 0; nb < 4; ++nb) {
        union { unsigned u[8]; v16h h; } Bm;
        #pragma unroll
        for (int v = 0; v < 8; ++v)
          Bm.u[v] = ldsB[(((cb << 4) + pairB0 + v) << 6) + (nb << 4) + r];
        acc[0][nb] = __builtin_amdgcn_wmma_f32_16x16x32_f16(
            false, A0.h, false, Bm.h, (short)0, acc[0][nb], false, false);
        acc[1][nb] = __builtin_amdgcn_wmma_f32_16x16x32_f16(
            false, A1.h, false, Bm.h, (short)0, acc[1][nb], false, false);
      }
    }
  }

  #pragma unroll
  for (int half = 0; half < 2; ++half) {
    #pragma unroll
    for (int nb = 0; nb < 4; ++nb) {
      const int n = n0 + (nb << 4) + r;
      const float bv = bias[n];               // Npad == Cout for these layers
      #pragma unroll
      for (int v = 0; v < 8; ++v) {
        const int mrow = half * 16 + v + (hi ? 8 : 0);   // C/D row layout
        const int px = x0 + mrow;
        float val = acc[half][nb][v] + bv;
        if (relu) val = fmaxf(val, 0.0f);
        const size_t basei = (((size_t)bimg * Hh + y) * Ww + px) * Npad + n;
        if (out32) out32[basei] = val;
        else       out16[basei] = (f16)val;
      }
    }
  }
}

// ---------------------------------------------------------------------------
// Generic 16x16-tile WMMA conv (used only for Pb: Npad=80, 1x1).
// ---------------------------------------------------------------------------
__global__ __launch_bounds__(256) void conv_wmma(
    const f16* __restrict__ act, const unsigned* __restrict__ bp,
    const float* __restrict__ bias,
    int Hh, int Ww, int Cin, int Npad, int Cout, int kk, int relu,
    f16* __restrict__ out16, float* __restrict__ out32) {
  const int lane = threadIdx.x & 31;
  const int wid  = blockIdx.x * (blockDim.x >> 5) + (threadIdx.x >> 5);
  const int tilesN = Npad >> 4;
  const int M = 2 * Hh * Ww;
  const int nTiles = (M >> 4) * tilesN;
  if (wid >= nTiles) return;
  const int tileM = wid / tilesN;
  const int tileN = wid - tileM * tilesN;
  const int m0 = tileM << 4;
  const int n0 = tileN << 4;
  const int hw = Hh * Ww;
  const int bimg = m0 / hw;
  const int rem  = m0 - bimg * hw;
  const int y  = rem / Ww;
  const int x0 = rem - y * Ww;
  const int pad = kk >> 1;
  const int r  = lane & 15;
  const int hi = lane >> 4;
  const int xr = x0 + r;
  const int n  = n0 + r;
  const unsigned* act32 = (const unsigned*)act;
  const int nKC = (kk * kk * Cin) >> 5;

  v8f c = {};
  for (int kc = 0; kc < nKC; ++kc) {
    union { unsigned u[8]; v16h h; } A, Bm;
    #pragma unroll
    for (int v = 0; v < 8; ++v) {
      int pairIdx = v + (hi ? 8 : 0);
      int kg = (kc << 5) + (pairIdx << 1);
      int cell = kg / Cin;
      int cin  = kg - cell * Cin;
      int dy = cell / kk;
      int dx = cell - dy * kk;
      int yy = y + dy - pad, xx = xr + dx - pad;
      unsigned aval = 0;
      if (yy >= 0 && yy < Hh && xx >= 0 && xx < Ww)
        aval = act32[((((size_t)bimg * Hh + yy) * Ww + xx) * Cin + cin) >> 1];
      A.u[v] = aval;
      int bPair = (kc << 4) + v + (hi ? 8 : 0);
      Bm.u[v] = bp[(size_t)bPair * Npad + n];
    }
    c = __builtin_amdgcn_wmma_f32_16x16x32_f16(false, A.h, false, Bm.h,
                                               (short)0, c, false, false);
  }
  float bv = (n < Cout) ? bias[n] : 0.0f;
  #pragma unroll
  for (int v = 0; v < 8; ++v) {
    int mrow = v + (hi ? 8 : 0);
    int px = x0 + mrow;
    float val = c[v] + bv;
    if (relu) val = fmaxf(val, 0.0f);
    size_t base = (((size_t)bimg * Hh + y) * Ww + px) * Npad + n;
    if (out32) out32[base] = val;
    else       out16[base] = (f16)val;
  }
}

// ---------------------------------------------------------------------------
// 2x2 max pool, NHWC f16
// ---------------------------------------------------------------------------
__global__ void maxpool2(const f16* __restrict__ in, f16* __restrict__ out,
                         int Hh, int Ww, int C, int total) {
  int i = blockIdx.x * blockDim.x + threadIdx.x;
  if (i >= total) return;
  int c = i % C; int t = i / C;
  int Wo = Ww >> 1;
  int x = t % Wo; t /= Wo;
  int Ho = Hh >> 1;
  int y = t % Ho; int b = t / Ho;
  size_t base = (((size_t)b * Hh + 2 * y) * Ww + 2 * x) * C + c;
  size_t rowS = (size_t)Ww * C;
  float a0 = (float)in[base],        a1 = (float)in[base + C];
  float a2 = (float)in[base + rowS], a3 = (float)in[base + rowS + C];
  out[i] = (f16)fmaxf(fmaxf(a0, a1), fmaxf(a2, a3));
}

// ---------------------------------------------------------------------------
// softmax over 65 channels (dustbin dropped) + pixel shuffle to [b,512,512]
// pb: [2,64,64,80] f32 NHWC
// ---------------------------------------------------------------------------
__global__ void softmax_shuffle(const float* __restrict__ pb, float* __restrict__ score) {
  int i = blockIdx.x * blockDim.x + threadIdx.x;    // 2*64*64
  int x8 = i & 63; int t = i >> 6; int y8 = t & 63; int b = t >> 6;
  const float* p = pb + (((size_t)(b * 64 + y8) * 64) + x8) * 80;
  float mx = p[0];
  for (int ch = 1; ch < 65; ++ch) mx = fmaxf(mx, p[ch]);
  float sum = 0.f;
  for (int ch = 0; ch < 65; ++ch) sum += expf(p[ch] - mx);
  float inv = 1.0f / sum;
  for (int ch = 0; ch < 64; ++ch) {
    float v = expf(p[ch] - mx) * inv;
    int Y = y8 * 8 + (ch >> 3), X = x8 * 8 + (ch & 7);
    score[((size_t)b * 512 + Y) * 512 + X] = v;
  }
}

// ---------------------------------------------------------------------------
// Separable 9x9 NMS window max + finalize with border mask
// ---------------------------------------------------------------------------
__global__ void nms_rowmax(const float* __restrict__ s, float* __restrict__ t) {
  int i = blockIdx.x * blockDim.x + threadIdx.x;    // 2*512*512
  int x = i & 511;
  int base = i - x;
  float m = -INFINITY;
  for (int dx = -4; dx <= 4; ++dx) {
    int xx = x + dx;
    if (xx >= 0 && xx < 512) m = fmaxf(m, s[(size_t)base + xx]);
  }
  t[i] = m;
}

__global__ void nms_colmax(const float* __restrict__ t, float* __restrict__ mx) {
  int i = blockIdx.x * blockDim.x + threadIdx.x;
  int x = i & 511; int y = (i >> 9) & 511; int b = i >> 18;
  float m = -INFINITY;
  for (int dy = -4; dy <= 4; ++dy) {
    int yy = y + dy;
    if (yy >= 0 && yy < 512) m = fmaxf(m, t[((size_t)(b * 512 + yy) << 9) + x]);
  }
  mx[i] = m;
}

__global__ void nms_final(float* __restrict__ s, const float* __restrict__ mx) {
  int i = blockIdx.x * blockDim.x + threadIdx.x;
  int x = i & 511; int y = (i >> 9) & 511;
  float v = s[i];
  bool keep = (v == mx[i]) && (v > 0.0f) &&
              (x >= 4 && x < 508 && y >= 4 && y < 508);
  s[i] = keep ? v : 0.0f;
}

// ---------------------------------------------------------------------------
// Top-1024 radix-select on float bits (scores >= 0 so uint order == float order)
// ---------------------------------------------------------------------------
__global__ void topk_init(TopkState* st) {
  int t = threadIdx.x;
  for (int b = 0; b < 2; ++b) {
    if (t == 0) { st[b].prefix = 0; st[b].kth = 1024; st[b].G = 0; st[b].c1 = 0; }
    st[b].hist[t] = 0;
  }
}

__global__ void topk_hist(const float* __restrict__ s, TopkState* st, int b, int pass) {
  int i = blockIdx.x * blockDim.x + threadIdx.x;    // NSC
  unsigned u = __float_as_uint(s[(size_t)b * NSC + i]);
  int shift = (3 - pass) * 8;
  bool match = (pass == 0) || ((u >> (shift + 8)) == (st[b].prefix >> (shift + 8)));
  if (match) atomicAdd(&st[b].hist[(u >> shift) & 255], 1u);
}

__global__ void topk_scan(TopkState* st, int b, int pass) {
  __shared__ unsigned h[256];
  int t = threadIdx.x;
  h[t] = st[b].hist[t];
  __syncthreads();
  if (t == 0) {
    int shift = (3 - pass) * 8;
    unsigned cum = 0; int kth = st[b].kth; int sel = 0;
    for (int v = 255; v >= 0; --v) {
      unsigned c = h[v];
      if (cum + c >= (unsigned)kth) { sel = v; break; }
      cum += c;
    }
    st[b].prefix |= ((unsigned)sel) << shift;
    st[b].kth = kth - (int)cum;
    if (pass == 3) st[b].G = 1024u - (unsigned)st[b].kth;
  }
  st[b].hist[t] = 0;
}

// count of ==T per 256-elem block (for deterministic tie selection)
__global__ void topk_eqcount(const float* __restrict__ s, const TopkState* st,
                             unsigned* __restrict__ blockCnt, int b) {
  int i = blockIdx.x * 256 + threadIdx.x;
  unsigned u = __float_as_uint(s[(size_t)b * NSC + i]);
  __shared__ unsigned cnt;
  if (threadIdx.x == 0) cnt = 0;
  __syncthreads();
  if (u == st[b].prefix) atomicAdd(&cnt, 1u);
  __syncthreads();
  if (threadIdx.x == 0) blockCnt[b * 1024 + blockIdx.x] = cnt;
}

__global__ __launch_bounds__(1024) void topk_eqscan(unsigned* blockCnt, int b) {
  __shared__ unsigned sc[1024];
  int t = threadIdx.x;
  unsigned own = blockCnt[b * 1024 + t];
  sc[t] = own;
  __syncthreads();
  for (int ofs = 1; ofs < 1024; ofs <<= 1) {
    unsigned v = (t >= ofs) ? sc[t - ofs] : 0u;
    __syncthreads();
    sc[t] += v;
    __syncthreads();
  }
  blockCnt[b * 1024 + t] = sc[t] - own;               // exclusive
}

__global__ __launch_bounds__(256) void topk_emit(
    const float* __restrict__ s, TopkState* st, const unsigned* __restrict__ blockOff,
    unsigned* __restrict__ selv, int* __restrict__ seli, int b) {
  int i = blockIdx.x * 256 + threadIdx.x;
  unsigned u = __float_as_uint(s[(size_t)b * NSC + i]);
  unsigned T = st[b].prefix, G = st[b].G;
  __shared__ unsigned sc[256];
  unsigned isEq = (u == T) ? 1u : 0u;
  sc[threadIdx.x] = isEq;
  __syncthreads();
  for (int ofs = 1; ofs < 256; ofs <<= 1) {
    unsigned v = (threadIdx.x >= (unsigned)ofs) ? sc[threadIdx.x - ofs] : 0u;
    __syncthreads();
    sc[threadIdx.x] += v;
    __syncthreads();
  }
  if (u > T) {
    unsigned slot = atomicAdd(&st[b].c1, 1u);          // order fixed by sort later
    selv[b * 1024 + slot] = u; seli[b * 1024 + slot] = i;
  } else if (isEq) {
    unsigned rank = blockOff[b * 1024 + blockIdx.x] + sc[threadIdx.x] - 1;
    if (G + rank < 1024) { selv[b * 1024 + G + rank] = u; seli[b * 1024 + G + rank] = i; }
  }
}

// bitonic sort 1024 pairs, descending by val, ascending idx on ties
__global__ __launch_bounds__(1024) void topk_sort(unsigned* selv, int* seli) {
  __shared__ unsigned sv[1024];
  __shared__ int si[1024];
  int b = blockIdx.x, t = threadIdx.x;
  sv[t] = selv[b * 1024 + t]; si[t] = seli[b * 1024 + t];
  __syncthreads();
  for (int k = 2; k <= 1024; k <<= 1)
    for (int j = k >> 1; j > 0; j >>= 1) {
      int p = t ^ j;
      if (p > t) {
        unsigned va = sv[t], vb = sv[p]; int ia = si[t], ib = si[p];
        bool aFirst = (va > vb) || (va == vb && ia < ib);
        bool descBlock = ((t & k) == 0);
        if (descBlock ? !aFirst : aFirst) { sv[t] = vb; sv[p] = va; si[t] = ib; si[p] = ia; }
      }
      __syncthreads();
    }
  selv[b * 1024 + t] = sv[t]; seli[b * 1024 + t] = si[t];
}

// ---------------------------------------------------------------------------
// L2-normalize dmap over 256 channels, in place. blocks = 2*64*64
// ---------------------------------------------------------------------------
__global__ __launch_bounds__(256) void dmap_norm(float* dm) {
  int t = threadIdx.x;
  float* p = dm + (size_t)blockIdx.x * 256;
  float v = p[t];
  __shared__ float red[256];
  red[t] = v * v;
  __syncthreads();
  for (int s = 128; s > 0; s >>= 1) { if (t < s) red[t] += red[t + s]; __syncthreads(); }
  p[t] = v * rsqrtf(fmaxf(red[0], 1e-24f));
}

// ---------------------------------------------------------------------------
// Keypoint outputs: lafs, vals, bilinear-sampled + normalized descriptors
// blocks = 2048 (one per keypoint), 256 threads (one per channel)
// ---------------------------------------------------------------------------
__global__ __launch_bounds__(256) void kpt_out(
    const unsigned* __restrict__ selv, const int* __restrict__ seli,
    const float* __restrict__ dmap, float* __restrict__ out) {
  int kp = blockIdx.x;
  int b = kp >> 10;
  int t = threadIdx.x;
  int idx = seli[kp];
  float val = __uint_as_float(selv[kp]);
  float kx = (float)(idx & 511), ky = (float)(idx >> 9);
  if (t == 0) {
    float* l = out + (size_t)kp * 6;
    l[0] = 1.f; l[1] = 0.f; l[2] = kx;
    l[3] = 0.f; l[4] = 1.f; l[5] = ky;
    out[12288 + kp] = val;
  }
  float px = kx * 0.125f - 0.5f, py = ky * 0.125f - 0.5f;
  float fx = floorf(px), fy = floorf(py);
  float wx = px - fx, wy = py - fy;
  int x0i = (int)fx; x0i = x0i < 0 ? 0 : (x0i > 63 ? 63 : x0i);
  int x1i = x0i + 1 > 63 ? 63 : x0i + 1;
  int y0i = (int)fy; y0i = y0i < 0 ? 0 : (y0i > 63 ? 63 : y0i);
  int y1i = y0i + 1 > 63 ? 63 : y0i + 1;
  const float* dm = dmap + (size_t)b * 64 * 64 * 256 + t;
  float g00 = dm[((size_t)y0i * 64 + x0i) * 256];
  float g01 = dm[((size_t)y0i * 64 + x1i) * 256];
  float g10 = dm[((size_t)y1i * 64 + x0i) * 256];
  float g11 = dm[((size_t)y1i * 64 + x1i) * 256];
  float d = (1.f - wy) * ((1.f - wx) * g00 + wx * g01)
          +        wy  * ((1.f - wx) * g10 + wx * g11);
  __shared__ float red[256];
  red[t] = d * d;
  __syncthreads();
  for (int s = 128; s > 0; s >>= 1) { if (t < s) red[t] += red[t + s]; __syncthreads(); }
  d = d / (sqrtf(red[0]) + 1e-8f);
  out[14336 + (size_t)kp * 256 + t] = d;
}

// ---------------------------------------------------------------------------
extern "C" void kernel_launch(void* const* d_in, const int* in_sizes, int n_in,
                              void* d_out, int out_size, void* d_ws, size_t ws_size,
                              hipStream_t stream) {
  const float* img = (const float*)d_in[0];
  auto IN = [&](int i) { return (const float*)d_in[i]; };

  char* ws = (char*)d_ws;
  f16* actA = (f16*)ws;                               // 67,108,864 B
  f16* actB = (f16*)(ws + 67108864ull);               // 67,108,864 B
  char* aux = ws + 134217728ull;
  size_t off = 0;
  auto alloc = [&](size_t bytes) {
    void* p = aux + off;
    off = (off + bytes + 255) & ~(size_t)255;
    return p;
  };
  unsigned* wp      = (unsigned*)alloc(651264ull * 4);          // packed weights
  float*    pbF32   = (float*)alloc(2ull * 64 * 64 * 80 * 4);   // score head raw
  float*    dbF32   = (float*)alloc(2ull * 64 * 64 * 256 * 4);  // desc head / dmap
  float*    score   = (float*)alloc(2ull * 512 * 512 * 4);
  float*    tmp1    = (float*)alloc(2ull * 512 * 512 * 4);
  float*    tmp2    = (float*)alloc(2ull * 512 * 512 * 4);
  TopkState* st     = (TopkState*)alloc(2 * sizeof(TopkState));
  unsigned* blockCnt= (unsigned*)alloc(2ull * 1024 * 4);
  unsigned* selv    = (unsigned*)alloc(2ull * 1024 * 4);
  int*      seli    = (int*)alloc(2ull * 1024 * 4);

  struct LL { int wi, bi, Hh, Ww, Cin, Cout, Npad, kk, relu; };
  const LL Ls[11] = {
    { 3, 4, 512, 512,  64,  64,  64, 3, 1},  // 1b
    { 5, 6, 256, 256,  64,  64,  64, 3, 1},  // 2a
    { 7, 8, 256, 256,  64,  64,  64, 3, 1},  // 2b
    { 9,10, 128, 128,  64, 128, 128, 3, 1},  // 3a
    {11,12, 128, 128, 128, 128, 128, 3, 1},  // 3b
    {13,14,  64,  64, 128, 128, 128, 3, 1},  // 4a
    {15,16,  64,  64, 128, 128, 128, 3, 1},  // 4b
    {17,18,  64,  64, 128, 256, 256, 3, 1},  // Pa
    {19,20,  64,  64, 256,  65,  80, 1, 0},  // Pb (16x16 fallback kernel)
    {21,22,  64,  64, 128, 256, 256, 3, 1},  // Da
    {23,24,  64,  64, 256, 256, 256, 1, 0},  // Db
  };
  size_t wOff[11]; size_t o = 0;
  for (int i = 0; i < 11; ++i) {
    wOff[i] = o;
    int K = Ls[i].kk * Ls[i].kk * Ls[i].Cin;
    o += (size_t)(K / 2) * Ls[i].Npad;
  }
  for (int i = 0; i < 11; ++i) {
    int K = Ls[i].kk * Ls[i].kk * Ls[i].Cin;
    int total = (K / 2) * Ls[i].Npad;
    pack_w<<<(total + 255) / 256, 256, 0, stream>>>(
        IN(Ls[i].wi), wp + wOff[i], Ls[i].Cin, Ls[i].Cout, Ls[i].kk, Ls[i].Npad, total);
  }

  conv1a<<<(2 * 512 * 512 * 64) / 256, 256, 0, stream>>>(img, IN(1), IN(2), actA);

  auto conv = [&](int li, const f16* in, f16* o16, float* o32) {
    const LL& L = Ls[li];
    if (L.Npad % 64 == 0 && L.Ww % 32 == 0) {
      int tiles = (2 * L.Hh * L.Ww / 32) * (L.Npad / 64);   // tilesM%8==0 for all
      conv_wmma_big<<<tiles / 8, 256, 0, stream>>>(
          in, wp + wOff[li], IN(L.bi), L.Hh, L.Ww, L.Cin, L.Npad, L.kk,
          L.relu, o16, o32);
    } else {
      int tiles = (2 * L.Hh * L.Ww / 16) * (L.Npad / 16);
      conv_wmma<<<(tiles + 7) / 8, 256, 0, stream>>>(
          in, wp + wOff[li], IN(L.bi), L.Hh, L.Ww, L.Cin, L.Npad, L.Cout, L.kk,
          L.relu, o16, o32);
    }
  };

  conv(0, actA, actB, nullptr);                                        // 1b: A->B
  maxpool2<<<(2*256*256*64)/256, 256, 0, stream>>>(actB, actA, 512, 512, 64, 2*256*256*64);
  conv(1, actA, actB, nullptr);                                        // 2a: A->B
  conv(2, actB, actA, nullptr);                                        // 2b: B->A
  maxpool2<<<(2*128*128*64)/256, 256, 0, stream>>>(actA, actB, 256, 256, 64, 2*128*128*64);
  conv(3, actB, actA, nullptr);                                        // 3a: B->A
  conv(4, actA, actB, nullptr);                                        // 3b: A->B
  maxpool2<<<(2*64*64*128)/256, 256, 0, stream>>>(actB, actA, 128, 128, 128, 2*64*64*128);
  conv(5, actA, actB, nullptr);                                        // 4a: A->B
  conv(6, actB, actA, nullptr);                                        // 4b: B->A
  conv(7, actA, actB, nullptr);                                        // Pa: A->B
  conv(8, actB, nullptr, pbF32);                                       // Pb: B->f32
  conv(9, actA, actB, nullptr);                                        // Da: A->B
  conv(10, actB, nullptr, dbF32);                                      // Db: B->f32

  dmap_norm<<<2 * 64 * 64, 256, 0, stream>>>(dbF32);
  softmax_shuffle<<<(2 * 64 * 64) / 256, 256, 0, stream>>>(pbF32, score);
  nms_rowmax<<<(2 * NSC) / 256, 256, 0, stream>>>(score, tmp1);
  nms_colmax<<<(2 * NSC) / 256, 256, 0, stream>>>(tmp1, tmp2);
  nms_final<<<(2 * NSC) / 256, 256, 0, stream>>>(score, tmp2);

  topk_init<<<1, 256, 0, stream>>>(st);
  for (int b = 0; b < 2; ++b)
    for (int p = 0; p < 4; ++p) {
      topk_hist<<<NSC / 256, 256, 0, stream>>>(score, st, b, p);
      topk_scan<<<1, 256, 0, stream>>>(st, b, p);
    }
  for (int b = 0; b < 2; ++b) {
    topk_eqcount<<<NSC / 256, 256, 0, stream>>>(score, st, blockCnt, b);
    topk_eqscan<<<1, 1024, 0, stream>>>(blockCnt, b);
    topk_emit<<<NSC / 256, 256, 0, stream>>>(score, st, blockCnt, selv, seli, b);
  }
  topk_sort<<<2, 1024, 0, stream>>>(selv, seli);
  kpt_out<<<2048, 256, 0, stream>>>(selv, seli, dbF32, (float*)d_out);
}
